// AttBlock_83210696393001
// MI455X (gfx1250) — matrix operand order repaired and checked by reading, not compiled, not run on hardware
//
#include <hip/hip_runtime.h>

// ---------------- problem constants (match reference) ----------------
#define N_NODES 100000
#define N_EDGES 20000
#define NNZ     500000
#define CH      256          // IN_CH == HID_CH == OUT_CH
#define N_HEADS 4
#define HEAD_DIM 64
#define NEG_SLOPE 0.2f
#define LN_EPS 1e-6f

// ---------------- workspace layout (bytes) ----------------
#define OFF_WP1   ((size_t)0)                         // 131072
#define OFF_WP2   ((size_t)131072)                    // 131072
#define OFF_XH    ((size_t)262144)                    // 102,400,000  (also AGG)
#define OFF_AGG   OFF_XH
#define OFF_ALPHA ((size_t)102662144)                 // 320,000
#define OFF_EX    ((size_t)102982144)                 // 8,000,000
#define OFF_XG    ((size_t)110982144)                 // 51,200,000 (bf16)
#define OFF_ESUM  ((size_t)162182144)                 // 20,480,000
#define OFF_CNT   ((size_t)182662144)                 // 80,000
#define OFF_DEN   ((size_t)182742144)                 // 1,600,000
#define OFF_M     ((size_t)184342144)                 // 1,600,000 (int)
#define ZERO_REGION_BYTES ((size_t)(20480000 + 80000 + 1600000))  // esum+cnt+den

// ---------------- vector types ----------------
typedef __attribute__((ext_vector_type(16))) __bf16          v16bf;
typedef __attribute__((ext_vector_type(8)))  float           v8f;
typedef __attribute__((ext_vector_type(16))) unsigned short  v16u;
typedef __attribute__((ext_vector_type(8)))  unsigned short  v8u;

__device__ __forceinline__ unsigned short f2bf(float f) {
    unsigned u = __float_as_uint(f);
    unsigned r = u + 0x7fffu + ((u >> 16) & 1u);   // round-to-nearest-even
    return (unsigned short)(r >> 16);
}
__device__ __forceinline__ int f2ord(float f) {
    int i = __float_as_int(f);
    return (i >= 0) ? i : (i ^ 0x7fffffff);        // order-preserving int key
}
__device__ __forceinline__ float ord2f(int i) {
    int j = (i >= 0) ? i : (i ^ 0x7fffffff);
    return __int_as_float(j);
}

// A-fragment load for fp32 source: per-lane halves are K = kb..kb+7, kb+16..kb+23
__device__ __forceinline__ v16bf load_a_f32(const float* __restrict__ xr, int kb) {
    float4 f0 = *(const float4*)(xr + kb);
    float4 f1 = *(const float4*)(xr + kb + 4);
    float4 f2 = *(const float4*)(xr + kb + 16);
    float4 f3 = *(const float4*)(xr + kb + 20);
    v16bf a;
    a[0]=(__bf16)f0.x;  a[1]=(__bf16)f0.y;  a[2]=(__bf16)f0.z;  a[3]=(__bf16)f0.w;
    a[4]=(__bf16)f1.x;  a[5]=(__bf16)f1.y;  a[6]=(__bf16)f1.z;  a[7]=(__bf16)f1.w;
    a[8]=(__bf16)f2.x;  a[9]=(__bf16)f2.y;  a[10]=(__bf16)f2.z; a[11]=(__bf16)f2.w;
    a[12]=(__bf16)f3.x; a[13]=(__bf16)f3.y; a[14]=(__bf16)f3.z; a[15]=(__bf16)f3.w;
    return a;
}

// A-fragment load for bf16 source
__device__ __forceinline__ v16bf load_a_bf16(const unsigned short* __restrict__ xr, int kb) {
    v8u lo = *(const v8u*)(xr + kb);
    v8u hi = *(const v8u*)(xr + kb + 16);
    v16u au = __builtin_shufflevector(lo, hi, 0,1,2,3,4,5,6,7,8,9,10,11,12,13,14,15);
    return __builtin_bit_cast(v16bf, au);
}

// ---------------- 0. init M to -inf-key ----------------
__global__ void init_m_kernel(int* __restrict__ M, int n) {
    int i = blockIdx.x * blockDim.x + threadIdx.x;
    if (i < n) M[i] = (int)0x80000000;  // INT_MIN: below every f2ord(float)
}

// ---------------- 1. pack weights to bf16 B-fragment layout ----------------
// Element (K,col) of [256x256] K-major weight -> p = (((K>>5)*256+col)*2 + ((K>>3)&1))*16 + ((K&7)|((K>>4)&1)<<3)
// so each lane's 16 contiguous halves are K = kt*32 + laneHi*8 + {0..7, 16..23} at one column.
__global__ void pack_kernel(const float* __restrict__ theta_w,   // [4,256,64]
                            const float* __restrict__ conv_w,    // [256,256]
                            unsigned short* __restrict__ Wp1,
                            unsigned short* __restrict__ Wp2) {
    int idx = blockIdx.x * 256 + threadIdx.x;      // 0..65535
    int K   = idx >> 8;
    int col = idx & 255;
    int kt = K >> 5, laneHi = (K >> 3) & 1, kk = (K & 7) | (((K >> 4) & 1) << 3);
    size_t p = (((size_t)kt * 256 + col) * 2 + laneHi) * 16 + kk;
    int h = col >> 6, d = col & 63;
    Wp1[p] = f2bf(theta_w[(h * 256 + K) * 64 + d]);
    Wp2[p] = f2bf(conv_w[K * 256 + col]);
}

// ---------------- 2. GEMM1: Xh = X @ W1 + theta_b (bf16 WMMA) ----------------
// Wave computes 16x64 tile; double-buffered A/B fragments across the K loop.
__global__ void __launch_bounds__(256)
gemm1_kernel(const float* __restrict__ X,
             const unsigned short* __restrict__ Wp,
             const float* __restrict__ bias,      // theta_b flat [256]
             float* __restrict__ Xh) {
    int tid  = threadIdx.x;
    int lane = tid & 31;
    int wave = tid >> 5;
    int mtile = blockIdx.x * 2 + (wave >> 2);
    int nt    = wave & 3;
    int laneLo = lane & 15;
    int laneHi = lane >> 4;
    const float* xr = X + (size_t)(mtile * 16 + laneLo) * CH;
    // per-lane B base: column nt*64+laneLo, K-half laneHi; kt stride 8192 halves, frag stride 512
    const unsigned short* bb =
        Wp + (((size_t)(nt * 64 + laneLo)) * 2 + laneHi) * 16;

    v8f acc[4] = {};
    v16bf a_cur = load_a_f32(xr, laneHi * 8);
    v16u  b_cur[4];
    #pragma unroll
    for (int f = 0; f < 4; ++f) b_cur[f] = *(const v16u*)(bb + f * 512);

    #pragma unroll
    for (int kt = 0; kt < 8; ++kt) {
        v16bf a_nxt = a_cur;
        v16u  b_nxt[4] = { b_cur[0], b_cur[1], b_cur[2], b_cur[3] };
        if (kt < 7) {
            a_nxt = load_a_f32(xr, (kt + 1) * 32 + laneHi * 8);
            #pragma unroll
            for (int f = 0; f < 4; ++f)
                b_nxt[f] = *(const v16u*)(bb + (size_t)(kt + 1) * 8192 + f * 512);
        }
        #pragma unroll
        for (int f = 0; f < 4; ++f)
            acc[f] = __builtin_amdgcn_wmma_f32_16x16x32_bf16(
                false, a_cur, false, __builtin_bit_cast(v16bf, b_cur[f]),
                (short)0, acc[f], false, false);
        a_cur = a_nxt;
        #pragma unroll
        for (int f = 0; f < 4; ++f) b_cur[f] = b_nxt[f];
    }

    int rbase = mtile * 16 + laneHi * 8;          // C layout: VGPR r -> row rbase+r
    #pragma unroll
    for (int f = 0; f < 4; ++f) {
        int col = nt * 64 + f * 16 + laneLo;
        float bv = bias[col];
        #pragma unroll
        for (int r = 0; r < 8; ++r)
            Xh[(size_t)(rbase + r) * CH + col] = acc[f][r] + bv;
    }
}

// ---------------- 3. v2e scatter: esum[e] += Xh[v], cnt[e] += 1 ----------------
__global__ void __launch_bounds__(256)
v2e_kernel(const int* __restrict__ eidx, const int* __restrict__ vidx,
           const float* __restrict__ Xh,
           float* __restrict__ esum, float* __restrict__ cnt) {
    int i = blockIdx.x;
    int t = threadIdx.x;
    int e = eidx[i], v = vidx[i];
    atomicAdd(&esum[(size_t)e * CH + t], Xh[(size_t)v * CH + t]);
    if (t == 0) atomicAdd(&cnt[e], 1.0f);
}

// ---------------- 4. edge finalize: Y = esum/max(cnt,1); alpha[e][h]=Y.aw -----
__global__ void __launch_bounds__(256)
edge_fin_kernel(float* __restrict__ esum,          // in-place -> Y
                const float* __restrict__ cnt,
                const float* __restrict__ atw,     // atten_w flat [256]
                float* __restrict__ alpha) {       // [E,4]
    __shared__ float s[256];
    int e = blockIdx.x, t = threadIdx.x;
    float c = fmaxf(cnt[e], 1.0f);
    float y = esum[(size_t)e * CH + t] / c;
    esum[(size_t)e * CH + t] = y;
    s[t] = y * atw[t];
    __syncthreads();
    #pragma unroll
    for (int off = 32; off > 0; off >>= 1) {
        if ((t & 63) < off) s[t] += s[t + off];
        __syncthreads();
    }
    if ((t & 63) == 0) alpha[e * N_HEADS + (t >> 6)] = s[t];
}

// ---------------- 5. softmax pass 1: segment max over vertices ----------------
__global__ void smax_kernel(const int* __restrict__ eidx, const int* __restrict__ vidx,
                            const float* __restrict__ alpha, int* __restrict__ M) {
    int idx = blockIdx.x * blockDim.x + threadIdx.x;
    if (idx >= NNZ * N_HEADS) return;
    int i = idx >> 2, h = idx & 3;
    float a = alpha[eidx[i] * N_HEADS + h];
    float sc = (a > 0.f) ? a : NEG_SLOPE * a;      // LeakyReLU
    atomicMax(&M[vidx[i] * N_HEADS + h], f2ord(sc));
}

// ---------------- 6. softmax pass 2: ex = exp(s-m), den += ex ----------------
__global__ void sexp_kernel(const int* __restrict__ eidx, const int* __restrict__ vidx,
                            const float* __restrict__ alpha, const int* __restrict__ M,
                            float* __restrict__ EX, float* __restrict__ den) {
    int idx = blockIdx.x * blockDim.x + threadIdx.x;
    if (idx >= NNZ * N_HEADS) return;
    int i = idx >> 2, h = idx & 3;
    int v = vidx[i];
    float a = alpha[eidx[i] * N_HEADS + h];
    float sc = (a > 0.f) ? a : NEG_SLOPE * a;
    float ex = expf(sc - ord2f(M[v * N_HEADS + h]));
    EX[idx] = ex;
    atomicAdd(&den[v * N_HEADS + h], ex);
}

// ---------------- 7. e2v weighted scatter: agg[v] += w * Y[e] ----------------
__global__ void __launch_bounds__(256)
e2v_kernel(const int* __restrict__ eidx, const int* __restrict__ vidx,
           const float* __restrict__ Y, const float* __restrict__ EX,
           const float* __restrict__ den, float* __restrict__ agg) {
    int i = blockIdx.x;
    int t = threadIdx.x;
    int e = eidx[i], v = vidx[i];
    int h = t >> 6;
    float w = EX[i * N_HEADS + h] / fmaxf(den[v * N_HEADS + h], 1e-12f);
    atomicAdd(&agg[(size_t)v * CH + t], w * Y[(size_t)e * CH + t]);
}

// ---------------- 8. fused ELU + LayerNorm + GELU -> bf16 Xg ----------------
__global__ void __launch_bounds__(256)
node_kernel(const float* __restrict__ agg,
            const float* __restrict__ ln_w, const float* __restrict__ ln_b,
            unsigned short* __restrict__ Xg) {
    __shared__ float red[256];
    int v = blockIdx.x, t = threadIdx.x;
    float x = agg[(size_t)v * CH + t];
    x = (x > 0.f) ? x : (expf(x) - 1.0f);          // ELU
    red[t] = x;  __syncthreads();
    #pragma unroll
    for (int off = 128; off > 0; off >>= 1) {
        if (t < off) red[t] += red[t + off];
        __syncthreads();
    }
    float mu = red[0] * (1.0f / CH);
    __syncthreads();
    float d = x - mu;
    red[t] = d * d;  __syncthreads();
    #pragma unroll
    for (int off = 128; off > 0; off >>= 1) {
        if (t < off) red[t] += red[t + off];
        __syncthreads();
    }
    float var = red[0] * (1.0f / CH);
    float xn = d * rsqrtf(var + LN_EPS) * ln_w[t] + ln_b[t];
    float g = 0.5f * xn * (1.0f + erff(xn * 0.70710678118654752f));  // exact GELU
    Xg[(size_t)v * CH + t] = f2bf(g);
}

// ------- 9. GEMM2: out = X + gamma * (Xg @ W2 + conv_b)  (bf16 WMMA) -------
__global__ void __launch_bounds__(256)
gemm2_kernel(const unsigned short* __restrict__ Xg,
             const unsigned short* __restrict__ Wp,
             const float* __restrict__ conv_b,
             const float* __restrict__ gamma,
             const float* __restrict__ Xin,
             float* __restrict__ out) {
    int tid  = threadIdx.x;
    int lane = tid & 31;
    int wave = tid >> 5;
    int mtile = blockIdx.x * 2 + (wave >> 2);
    int nt    = wave & 3;
    int laneLo = lane & 15;
    int laneHi = lane >> 4;
    const unsigned short* xr = Xg + (size_t)(mtile * 16 + laneLo) * CH;
    const unsigned short* bb =
        Wp + (((size_t)(nt * 64 + laneLo)) * 2 + laneHi) * 16;

    v8f acc[4] = {};
    v16bf a_cur = load_a_bf16(xr, laneHi * 8);
    v16u  b_cur[4];
    #pragma unroll
    for (int f = 0; f < 4; ++f) b_cur[f] = *(const v16u*)(bb + f * 512);

    #pragma unroll
    for (int kt = 0; kt < 8; ++kt) {
        v16bf a_nxt = a_cur;
        v16u  b_nxt[4] = { b_cur[0], b_cur[1], b_cur[2], b_cur[3] };
        if (kt < 7) {
            a_nxt = load_a_bf16(xr, (kt + 1) * 32 + laneHi * 8);
            #pragma unroll
            for (int f = 0; f < 4; ++f)
                b_nxt[f] = *(const v16u*)(bb + (size_t)(kt + 1) * 8192 + f * 512);
        }
        #pragma unroll
        for (int f = 0; f < 4; ++f)
            acc[f] = __builtin_amdgcn_wmma_f32_16x16x32_bf16(
                false, a_cur, false, __builtin_bit_cast(v16bf, b_cur[f]),
                (short)0, acc[f], false, false);
        a_cur = a_nxt;
        #pragma unroll
        for (int f = 0; f < 4; ++f) b_cur[f] = b_nxt[f];
    }

    int rbase = mtile * 16 + laneHi * 8;
    #pragma unroll
    for (int f = 0; f < 4; ++f) {
        int col = nt * 64 + f * 16 + laneLo;
        float bv = conv_b[col];
        float gv = gamma[col];
        #pragma unroll
        for (int r = 0; r < 8; ++r) {
            size_t o = (size_t)(rbase + r) * CH + col;
            out[o] = Xin[o] + gv * (acc[f][r] + bv);
        }
    }
}

// ---------------- launcher ----------------
extern "C" void kernel_launch(void* const* d_in, const int* in_sizes, int n_in,
                              void* d_out, int out_size, void* d_ws, size_t ws_size,
                              hipStream_t stream) {
    const float* X       = (const float*)d_in[0];
    const int*   eidx    = (const int*)d_in[1];
    const int*   vidx    = (const int*)d_in[2];
    const float* theta_w = (const float*)d_in[3];
    const float* theta_b = (const float*)d_in[4];   // flat [256] == col order
    const float* atten_w = (const float*)d_in[5];   // flat [256] == channel order
    const float* ln_w    = (const float*)d_in[6];
    const float* ln_b    = (const float*)d_in[7];
    const float* conv_w  = (const float*)d_in[8];
    const float* conv_b  = (const float*)d_in[9];
    const float* gamma   = (const float*)d_in[10];
    float* out = (float*)d_out;

    char* ws = (char*)d_ws;
    unsigned short* WP1 = (unsigned short*)(ws + OFF_WP1);
    unsigned short* WP2 = (unsigned short*)(ws + OFF_WP2);
    float* XH    = (float*)(ws + OFF_XH);
    float* AGG   = (float*)(ws + OFF_AGG);          // aliases XH (rezeroed)
    float* ALPHA = (float*)(ws + OFF_ALPHA);
    float* EX    = (float*)(ws + OFF_EX);
    unsigned short* XG = (unsigned short*)(ws + OFF_XG);
    float* ESUM  = (float*)(ws + OFF_ESUM);
    float* CNT   = (float*)(ws + OFF_CNT);
    float* DEN   = (float*)(ws + OFF_DEN);
    int*   M     = (int*)(ws + OFF_M);

    // zero esum/cnt/den (contiguous)
    hipMemsetAsync(ws + OFF_ESUM, 0, ZERO_REGION_BYTES, stream);
    init_m_kernel<<<(N_NODES * N_HEADS + 255) / 256, 256, 0, stream>>>(M, N_NODES * N_HEADS);
    pack_kernel<<<256, 256, 0, stream>>>(theta_w, conv_w, WP1, WP2);

    gemm1_kernel<<<N_NODES / 32, 256, 0, stream>>>(X, WP1, theta_b, XH);

    v2e_kernel<<<NNZ, 256, 0, stream>>>(eidx, vidx, XH, ESUM, CNT);

    // XH no longer needed -> reuse as AGG (zero it)
    hipMemsetAsync(ws + OFF_AGG, 0, (size_t)N_NODES * CH * sizeof(float), stream);

    edge_fin_kernel<<<N_EDGES, 256, 0, stream>>>(ESUM, CNT, atten_w, ALPHA);
    smax_kernel<<<(NNZ * N_HEADS + 255) / 256, 256, 0, stream>>>(eidx, vidx, ALPHA, M);
    sexp_kernel<<<(NNZ * N_HEADS + 255) / 256, 256, 0, stream>>>(eidx, vidx, ALPHA, M, EX, DEN);
    e2v_kernel<<<NNZ, 256, 0, stream>>>(eidx, vidx, ESUM, EX, DEN, AGG);

    node_kernel<<<N_NODES, 256, 0, stream>>>(AGG, ln_w, ln_b, XG);

    gemm2_kernel<<<N_NODES / 32, 256, 0, stream>>>(XG, WP2, conv_b, gamma, X, out);
}
